// MSCLCMI_82171314307449
// MI455X (gfx1250) — matrix operand order repaired and checked by reading, not compiled, not run on hardware
//
#include <hip/hip_runtime.h>

// ---------------------------------------------------------------------------
// Types for WMMA (gfx1250, wave32)
// ---------------------------------------------------------------------------
typedef __attribute__((ext_vector_type(16))) __bf16    bf16x16;
typedef __attribute__((ext_vector_type(8)))  float     f32x8;
typedef __attribute__((ext_vector_type(4)))  unsigned  u32x4;

union BFrag { bf16x16 v; u32x4 q[2]; };

#define M_MOD 4
#define DIMD  256
#define DIMP  256

// round-to-nearest-even f32 -> bf16 (stored as u16)
static __device__ inline unsigned short f32_to_bf16(float f) {
    unsigned int u = __float_as_uint(f);
    unsigned int r = (u + 0x7FFFu + ((u >> 16) & 1u)) >> 16;
    return (unsigned short)r;
}

// ---------------------------------------------------------------------------
// CDNA5 async copy global -> LDS (ASYNCcnt-tracked bulk staging)
// lds_byte_addr is the workgroup-relative LDS byte offset (dynamic LDS starts
// at offset 0 when no static LDS is allocated).
// ---------------------------------------------------------------------------
static __device__ inline void async_copy_b128(unsigned lds_byte_addr, const void* gptr) {
    unsigned long long ga = (unsigned long long)(uintptr_t)gptr;
    asm volatile("global_load_async_to_lds_b128 %0, %1, off"
                 :: "v"(lds_byte_addr), "v"(ga)
                 : "memory");
}
static __device__ inline void wait_async0() {
    asm volatile("s_wait_asynccnt 0" ::: "memory");
}

// A fragment: 16x32 bf16, row-major source, lane m = lane&15, g = lane>>4
// elements e=0..7  -> k = kBase + 8*g + e
// elements e=8..15 -> k = kBase + 16 + 8*g + (e-8)
static __device__ inline bf16x16 load_a_frag(const unsigned short* base, int row,
                                             int ldk, int kBase, int lane) {
    int g = lane >> 4;
    const unsigned short* p = base + (size_t)row * ldk + kBase + 8 * g;
    BFrag f;
    f.q[0] = *(const u32x4*)(p);
    f.q[1] = *(const u32x4*)(p + 16);
    return f.v;
}

// B fragment: 32x16 bf16 taken from a TRANSPOSED weight (WT[n][k]), lane n = lane&15
// elements e=0..7  -> k = kBase + 8*g + e
// elements e=8..15 -> k = kBase + 16 + 8*g + (e-8)
static __device__ inline bf16x16 load_b_frag(const unsigned short* base, int col,
                                             int ldk, int kBase, int lane) {
    int g = lane >> 4;
    const unsigned short* p = base + (size_t)col * ldk + kBase + 8 * g;
    BFrag f;
    f.q[0] = *(const u32x4*)(p);
    f.q[1] = *(const u32x4*)(p + 16);
    return f.v;
}

static __device__ inline void atomicMaxF(float* addr, float val) {
    if (val >= 0.0f) atomicMax((int*)addr, __float_as_int(val));
    else             atomicMin((unsigned int*)addr, __float_as_uint(val));
}

// ---------------------------------------------------------------------------
// Conversion kernels
// ---------------------------------------------------------------------------
__global__ void cvt_bf16_kernel(const float* __restrict__ src,
                                unsigned short* __restrict__ dst, int total) {
    int i = blockIdx.x * blockDim.x + threadIdx.x;
    if (i < total) dst[i] = f32_to_bf16(src[i]);
}

// src[m][k*C + c] -> dst[m][c*K + k]  (transpose + convert per modality)
__global__ void cvt_transpose_kernel(const float* __restrict__ src,
                                     unsigned short* __restrict__ dst,
                                     int K, int C, int total) {
    int i = blockIdx.x * blockDim.x + threadIdx.x;
    if (i >= total) return;
    int m = i / (K * C);
    int r = i - m * (K * C);
    int k = r / C;
    int c = r - k * C;
    dst[(size_t)m * K * C + (size_t)c * K + k] = f32_to_bf16(src[i]);
}

// ---------------------------------------------------------------------------
// Stage A: h = x @ Wg per modality (bf16 WMMA, f32 accumulate)
// grid ((N/16+7)/8, M), block 256 (8 waves), 128 KB dynamic LDS.
// Whole Wg^T[m] (256x256 bf16) is staged into LDS with async-to-LDS copies;
// each wave computes one 16-row strip, A fragments hoisted into registers.
// ---------------------------------------------------------------------------
__global__ void gemm_h_kernel(const unsigned short* __restrict__ X16,
                              const unsigned short* __restrict__ WgT16,
                              float* __restrict__ H, int N) {
    extern __shared__ unsigned short sB[];   // 256*256 bf16 = 128 KB
    int lane = threadIdx.x & 31;
    int wid  = threadIdx.x >> 5;
    int m    = blockIdx.y;
    const unsigned short* A  = X16   + (size_t)m * N * DIMD;
    const unsigned short* Bm = WgT16 + (size_t)m * DIMD * DIMD;
    float*                Hm = H     + (size_t)m * N * DIMD;

    // ---- async bulk stage: 32 iterations x 256 threads x 16B = 128 KB ----
    {
        const char* gsrc = (const char*)Bm;
#pragma unroll 4
        for (int it = 0; it < 32; ++it) {
            unsigned off = (unsigned)((it * 256 + threadIdx.x) * 16);
            async_copy_b128(off, gsrc + off);
        }
        wait_async0();
    }
    __syncthreads();

    int NT = N >> 4;
    int nodeTile = blockIdx.x * 8 + wid;
    if (nodeTile < NT) {
        int rowBase = nodeTile * 16;
        int aRow = rowBase + (lane & 15);
        int g    = lane >> 4;

        // hoist all 8 A fragments (K = 256) for this strip: loaded once
        bf16x16 aF[8];
#pragma unroll
        for (int kt = 0; kt < 8; ++kt)
            aF[kt] = load_a_frag(A, aRow, DIMD, kt * 32, lane);

#pragma unroll 1
        for (int ct = 0; ct < 16; ++ct) {
            int colBase = ct * 16;
            int bRow = colBase + (lane & 15);
            f32x8 acc = {0.f,0.f,0.f,0.f,0.f,0.f,0.f,0.f};
#pragma unroll
            for (int kt = 0; kt < 8; ++kt) {
                bf16x16 b = load_b_frag(sB, bRow, DIMD, kt * 32, lane);
                acc = __builtin_amdgcn_wmma_f32_16x16x32_bf16(
                          false, aF[kt], false, b, (short)0, acc, false, false);
            }
            int col = colBase + (lane & 15);
#pragma unroll
            for (int r = 0; r < 8; ++r)
                Hm[(size_t)(rowBase + r + 8 * g) * DIMD + col] = acc[r];
        }
    }
}

// ---------------------------------------------------------------------------
// Per-node attention logits: s[n] = h[n]·a_src, t[n] = h[n]·a_dst
// one wave per (m, n)
// ---------------------------------------------------------------------------
__global__ void st_kernel(const float* __restrict__ H,
                          const float* __restrict__ a_src,
                          const float* __restrict__ a_dst,
                          float* __restrict__ S, float* __restrict__ T, int N) {
    int lane = threadIdx.x & 31;
    int gw = blockIdx.x * 8 + (threadIdx.x >> 5);
    if (gw >= M_MOD * N) return;
    int m = gw / N;
    const float* h  = H + (size_t)gw * DIMD;
    const float* as = a_src + m * DIMD;
    const float* ad = a_dst + m * DIMD;
    float ps = 0.f, pt = 0.f;
#pragma unroll
    for (int j = 0; j < 8; ++j) {
        int d = lane + 32 * j;
        float hv = h[d];
        ps += hv * as[d];
        pt += hv * ad[d];
    }
#pragma unroll
    for (int off = 16; off > 0; off >>= 1) {
        ps += __shfl_xor(ps, off, 32);
        pt += __shfl_xor(pt, off, 32);
    }
    if (lane == 0) { S[gw] = ps; T[gw] = pt; }
}

// ---------------------------------------------------------------------------
// Init: zero AGG, mx = -inf, denom = 0
// ---------------------------------------------------------------------------
__global__ void init_kernel(float* __restrict__ AGG, float* __restrict__ MX,
                            float* __restrict__ DEN, int totalAgg, int totalMN) {
    int i = blockIdx.x * blockDim.x + threadIdx.x;
    if (i < totalAgg) AGG[i] = 0.f;
    if (i < totalMN) { MX[i] = __int_as_float(0xFF800000); DEN[i] = 0.f; }
}

// ---------------------------------------------------------------------------
// Edge pass 1: alpha = leakyrelu(s[src]+t[dst]); segment max
// ---------------------------------------------------------------------------
__global__ void edge1_kernel(const int* __restrict__ ei,
                             const float* __restrict__ S, const float* __restrict__ T,
                             float* __restrict__ ALPHA, float* __restrict__ MX,
                             int N, int E, int EN) {
    int i = blockIdx.x * blockDim.x + threadIdx.x;
    if (i >= M_MOD * EN) return;
    int m = i / EN, e = i - m * EN;
    int src = (e < E) ? ei[e]     : (e - E);
    int dst = (e < E) ? ei[E + e] : (e - E);
    float a = S[m * N + src] + T[m * N + dst];
    a = (a > 0.f) ? a : 0.2f * a;
    ALPHA[i] = a;
    atomicMaxF(&MX[m * N + dst], a);
}

// Edge pass 2: e = exp(alpha - mx[dst]); segment sum
__global__ void edge2_kernel(const int* __restrict__ ei,
                             float* __restrict__ ALPHA,
                             const float* __restrict__ MX, float* __restrict__ DEN,
                             int N, int E, int EN) {
    int i = blockIdx.x * blockDim.x + threadIdx.x;
    if (i >= M_MOD * EN) return;
    int m = i / EN, e = i - m * EN;
    int dst = (e < E) ? ei[E + e] : (e - E);
    float ev = __expf(ALPHA[i] - MX[m * N + dst]);
    ALPHA[i] = ev;
    atomicAdd(&DEN[m * N + dst], ev);
}

// Edge pass 3: AGG[dst] += e * h[src]   (one wave per (m, edge))
__global__ void edge3_kernel(const int* __restrict__ ei,
                             const float* __restrict__ ALPHA,
                             const float* __restrict__ H, float* __restrict__ AGG,
                             int N, int E, int EN) {
    int lane = threadIdx.x & 31;
    int gw = blockIdx.x * 8 + (threadIdx.x >> 5);
    if (gw >= M_MOD * EN) return;
    int m = gw / EN, e = gw - m * EN;
    int src = (e < E) ? ei[e]     : (e - E);
    int dst = (e < E) ? ei[E + e] : (e - E);
    float ev = ALPHA[gw];
    const float* hs = H   + ((size_t)m * N + src) * DIMD;
    float*       ag = AGG + ((size_t)m * N + dst) * DIMD;
#pragma unroll
    for (int j = 0; j < 8; ++j) {
        int d = lane + 32 * j;
        atomicAdd(&ag[d], ev * hs[d]);
    }
}

// ---------------------------------------------------------------------------
// Stage C: h_gat = relu(agg/denom + bg) -> bf16
// ---------------------------------------------------------------------------
__global__ void epilogue_kernel(const float* __restrict__ AGG,
                                const float* __restrict__ DEN,
                                const float* __restrict__ bg,
                                unsigned short* __restrict__ H2,
                                int N, int total) {
    int i = blockIdx.x * blockDim.x + threadIdx.x;
    if (i >= total) return;
    int d  = i & (DIMD - 1);
    int mn = i >> 8;          // m*N + n
    int m  = mn / N;
    float v = AGG[i] / (DEN[mn] + 1e-16f) + bg[m * DIMD + d];
    v = (v > 0.f) ? v : 0.f;
    H2[i] = f32_to_bf16(v);
}

// ---------------------------------------------------------------------------
// Stage D: out = sum_m conv_w[m] * (h_gat_m @ Wp_m) + sum_m conv_w[m]*bp_m + conv_b
// grid N/16, block 256; wave w covers col tiles 2w, 2w+1
// ---------------------------------------------------------------------------
__global__ void gemm_fuse_kernel(const unsigned short* __restrict__ H2,
                                 const unsigned short* __restrict__ WpT16,
                                 const float* __restrict__ conv_w,
                                 const float* __restrict__ conv_b,
                                 const float* __restrict__ bp,
                                 float* __restrict__ out, int N) {
    int lane = threadIdx.x & 31;
    int wid  = threadIdx.x >> 5;
    int rowBase = blockIdx.x * 16;
    int aRow = rowBase + (lane & 15);
    int g    = lane >> 4;

    for (int t = 0; t < 2; ++t) {
        int colBase = (wid * 2 + t) * 16;
        int bRow = colBase + (lane & 15);
        f32x8 tot = {0.f,0.f,0.f,0.f,0.f,0.f,0.f,0.f};
        for (int m = 0; m < M_MOD; ++m) {
            const unsigned short* A = H2    + (size_t)m * N * DIMD;
            const unsigned short* B = WpT16 + (size_t)m * DIMP * DIMD;
            f32x8 acc = {0.f,0.f,0.f,0.f,0.f,0.f,0.f,0.f};
#pragma unroll
            for (int kt = 0; kt < 8; ++kt) {
                __builtin_prefetch(A + (size_t)aRow * DIMD + ((kt + 1) & 7) * 32, 0, 1);
                bf16x16 a = load_a_frag(A, aRow, DIMD, kt * 32, lane);
                bf16x16 b = load_b_frag(B, bRow, DIMD, kt * 32, lane);
                acc = __builtin_amdgcn_wmma_f32_16x16x32_bf16(
                          false, a, false, b, (short)0, acc, false, false);
            }
            float cw = conv_w[m];
#pragma unroll
            for (int r = 0; r < 8; ++r) tot[r] += cw * acc[r];
        }
        int col = colBase + (lane & 15);
        float bias = conv_b[0];
#pragma unroll
        for (int m = 0; m < M_MOD; ++m) bias += conv_w[m] * bp[m * DIMP + col];
#pragma unroll
        for (int r = 0; r < 8; ++r)
            out[(size_t)(rowBase + r + 8 * g) * DIMP + col] = tot[r] + bias;
    }
}

// ---------------------------------------------------------------------------
// Host launcher
// ---------------------------------------------------------------------------
extern "C" void kernel_launch(void* const* d_in, const int* in_sizes, int n_in,
                              void* d_out, int out_size, void* d_ws, size_t ws_size,
                              hipStream_t stream) {
    const float* x      = (const float*)d_in[0];
    const int*   ei     = (const int*)  d_in[1];
    const float* Wg     = (const float*)d_in[2];
    const float* a_src  = (const float*)d_in[3];
    const float* a_dst  = (const float*)d_in[4];
    const float* bg     = (const float*)d_in[5];
    const float* Wp     = (const float*)d_in[6];
    const float* bp     = (const float*)d_in[7];
    const float* conv_w = (const float*)d_in[8];
    const float* conv_b = (const float*)d_in[9];
    float*       out    = (float*)d_out;

    const int M = M_MOD, D = DIMD, P = DIMP;
    const int N  = in_sizes[0] / (M * D);
    const int E  = in_sizes[1] / 2;
    const int EN = E + N;

    // carve workspace (256B aligned chunks)
    char* wp = (char*)d_ws;
    auto carve = [&](size_t bytes) -> char* {
        char* p = wp;
        wp += (bytes + 255) & ~(size_t)255;
        return p;
    };
    unsigned short* X16   = (unsigned short*)carve((size_t)M * N * D * 2);
    unsigned short* WgT16 = (unsigned short*)carve((size_t)M * D * D * 2);
    unsigned short* WpT16 = (unsigned short*)carve((size_t)M * D * P * 2);
    float*          H     = (float*)carve((size_t)M * N * D * 4);
    float*          S     = (float*)carve((size_t)M * N * 4);
    float*          T     = (float*)carve((size_t)M * N * 4);
    float*          ALPHA = (float*)carve((size_t)M * EN * 4);
    float*          MX    = (float*)carve((size_t)M * N * 4);
    float*          DEN   = (float*)carve((size_t)M * N * 4);
    float*          AGG   = (float*)carve((size_t)M * N * D * 4);
    unsigned short* H2    = (unsigned short*)carve((size_t)M * N * D * 2);

    const int threads = 256;
    int totalX   = M * N * D;
    int totalWg  = M * D * D;
    int totalWp  = M * D * P;
    int totalMN  = M * N;
    int totalME  = M * EN;
    int NT       = N / 16;

    // conversions
    cvt_bf16_kernel<<<(totalX + 255) / 256, threads, 0, stream>>>(x, X16, totalX);
    cvt_transpose_kernel<<<(totalWg + 255) / 256, threads, 0, stream>>>(Wg, WgT16, D, D, totalWg);
    cvt_transpose_kernel<<<(totalWp + 255) / 256, threads, 0, stream>>>(Wp, WpT16, D, P, totalWp);

    // h = x @ Wg (WMMA, Wg^T staged to LDS via async-to-LDS)
    gemm_h_kernel<<<dim3((NT + 7) / 8, M), threads, (size_t)D * D * 2, stream>>>(X16, WgT16, H, N);

    // attention logits
    st_kernel<<<(totalMN + 7) / 8, threads, 0, stream>>>(H, a_src, a_dst, S, T, N);

    // init scatter buffers
    init_kernel<<<(totalX + 255) / 256, threads, 0, stream>>>(AGG, MX, DEN, totalX, totalMN);

    // edge softmax + aggregate
    edge1_kernel<<<(totalME + 255) / 256, threads, 0, stream>>>(ei, S, T, ALPHA, MX, N, E, EN);
    edge2_kernel<<<(totalME + 255) / 256, threads, 0, stream>>>(ei, ALPHA, MX, DEN, N, E, EN);
    edge3_kernel<<<(totalME + 7) / 8, threads, 0, stream>>>(ei, ALPHA, H, AGG, N, E, EN);

    // relu epilogue -> bf16
    epilogue_kernel<<<(totalX + 255) / 256, threads, 0, stream>>>(AGG, DEN, bg, H2, N, totalX);

    // fused projection + channel mix (WMMA)
    gemm_fuse_kernel<<<N / 16, threads, 0, stream>>>(H2, WpT16, conv_w, conv_b, bp, out, N);

    (void)n_in; (void)out_size; (void)ws_size;
}
